// VL_SAE_16312285790427
// MI455X (gfx1250) — matrix operand order repaired and checked by reading, not compiled, not run on hardware
//
#include <hip/hip_runtime.h>

// Problem sizes from the reference
#define NN 4096
#define DD 768
#define HH 16384

typedef __attribute__((ext_vector_type(8)))  float  v8f;
typedef __attribute__((ext_vector_type(8)))  __bf16 v8bf;
typedef __attribute__((ext_vector_type(16))) __bf16 v16bf;

// ---------- helpers: float <-> bf16 bits (round-to-nearest-even) ----------
__device__ __forceinline__ unsigned short f32_to_bf16_rne(float f) {
    unsigned u = __float_as_uint(f);
    unsigned r = 0x7FFFu + ((u >> 16) & 1u);
    return (unsigned short)((u + r) >> 16);
}
__device__ __forceinline__ float bf16_bits_to_f32(unsigned short h) {
    return __uint_as_float(((unsigned)h) << 16);
}

// A fragment (16x32 bf16): per lane, one contiguous 16-elem K-chunk.
__device__ __forceinline__ v16bf load_frag_a(const __bf16* p) {
    v8bf c0 = *(const v8bf*)(p);
    v8bf c1 = *(const v8bf*)(p + 8);
    return __builtin_shufflevector(c0, c1, 0,1,2,3,4,5,6,7,8,9,10,11,12,13,14,15);
}
// B fragment (32x16 bf16): per lane, one contiguous 16-elem K-chunk.
__device__ __forceinline__ v16bf load_frag_b(const __bf16* p) {
    v8bf c0 = *(const v8bf*)(p);
    v8bf c1 = *(const v8bf*)(p + 8);
    return __builtin_shufflevector(c0, c1, 0,1,2,3,4,5,6,7,8,9,10,11,12,13,14,15);
}

// ---------------------------------------------------------------------------
// Kernel 1: row-wise L2 normalize + split into bf16 hi/lo residual pair.
// ---------------------------------------------------------------------------
__global__ __launch_bounds__(256) void l2norm_split_kernel(
    const float* __restrict__ in,
    unsigned short* __restrict__ out_hi,
    unsigned short* __restrict__ out_lo,
    int cols)
{
    __shared__ float red[8];
    const int row = blockIdx.x;
    const float* src = in + (size_t)row * cols;

    float ss = 0.0f;
    for (int i = threadIdx.x; i < cols; i += 256) { float v = src[i]; ss += v * v; }
    for (int off = 16; off > 0; off >>= 1) ss += __shfl_down(ss, off, 32);
    if ((threadIdx.x & 31) == 0) red[threadIdx.x >> 5] = ss;
    __syncthreads();
    if (threadIdx.x == 0) {
        float t = 0.0f;
        for (int i = 0; i < 8; ++i) t += red[i];
        red[0] = 1.0f / fmaxf(sqrtf(t), 1e-12f);
    }
    __syncthreads();
    const float inv = red[0];

    unsigned short* dh = out_hi + (size_t)row * cols;
    unsigned short* dl = out_lo + (size_t)row * cols;
    for (int i = threadIdx.x; i < cols; i += 256) {
        float x = src[i] * inv;
        unsigned short h = f32_to_bf16_rne(x);
        dh[i] = h;
        dl[i] = f32_to_bf16_rne(x - bf16_bits_to_f32(h));
    }
}

// ---------------------------------------------------------------------------
// Kernel 2: tiled transpose Wv/Wt [D,H] -> [H,D].
// ---------------------------------------------------------------------------
__global__ __launch_bounds__(256) void transpose_kernel(
    const float* __restrict__ in,   // [R, C] row-major (R=D, C=H)
    float* __restrict__ out,        // [C, R] row-major
    int R, int C)
{
    __shared__ float tile[32][33];
    const int c0 = blockIdx.x * 32;
    const int r0 = blockIdx.y * 32;
    const int tx = threadIdx.x & 31;
    const int ty = threadIdx.x >> 5;   // 0..7
    for (int i = ty; i < 32; i += 8)
        tile[i][tx] = in[(size_t)(r0 + i) * C + (c0 + tx)];
    __syncthreads();
    for (int i = ty; i < 32; i += 8)
        out[(size_t)(c0 + i) * R + (r0 + tx)] = tile[tx][i];
}

// ---------------------------------------------------------------------------
// Kernel 3: sim = 2 - sqrt(max(2 - 2*(e @ wN^T), 0)) via bf16x3 WMMA.
// Grid: (H/512, N/32, 2 modalities), block 256 = 8 waves.
// Each wave computes a 32(M) x 64(N) tile: two A fragment pairs share each
// B fragment pair -> 24 WMMAs per 24 b128 loads per K=32 step, and B-matrix
// L2 traffic is halved vs a 16-row M tile.
// cos = Ahi*Bhi + Ahi*Blo + Alo*Bhi  (lo*lo dropped, ~1e-7 error)
// ---------------------------------------------------------------------------
__global__ __launch_bounds__(256) void sim_gemm_kernel(
    const unsigned short* __restrict__ eVhi, const unsigned short* __restrict__ eVlo,
    const unsigned short* __restrict__ eThi, const unsigned short* __restrict__ eTlo,
    const unsigned short* __restrict__ wHi,  const unsigned short* __restrict__ wLo,
    float* __restrict__ latent_base)        // [2][N][H]
{
    const int lane = threadIdx.x & 31;
    const int wave = threadIdx.x >> 5;
    const int m0   = blockIdx.y * 32;
    const int h0   = blockIdx.x * 512 + wave * 64;
    const int half = lane >> 4;            // 0 or 1
    const int r    = lane & 15;

    const unsigned short* Ahi = blockIdx.z ? eThi : eVhi;
    const unsigned short* Alo = blockIdx.z ? eTlo : eVlo;
    float* latent = latent_base + (size_t)blockIdx.z * NN * HH;

    // A fragments: lane holds row (m0 [+16] + r); contiguous chunk at 16*half
    const __bf16* a0h = (const __bf16*)(Ahi + (size_t)(m0 + r) * DD + 16 * half);
    const __bf16* a0l = (const __bf16*)(Alo + (size_t)(m0 + r) * DD + 16 * half);
    const __bf16* a1h = a0h + (size_t)16 * DD;
    const __bf16* a1l = a0l + (size_t)16 * DD;
    // B fragments: lane holds encoder row (h0+16j+r); contiguous chunk at 16*half
    const size_t bOff = (size_t)(h0 + r) * DD + 16 * half;
    const __bf16* bHiP = (const __bf16*)wHi + bOff;
    const __bf16* bLoP = (const __bf16*)wLo + bOff;

    v8f acc0[4] = {};
    v8f acc1[4] = {};
    for (int k = 0; k < DD; k += 32) {
        v16bf A0h = load_frag_a(a0h + k);
        v16bf A0l = load_frag_a(a0l + k);
        v16bf A1h = load_frag_a(a1h + k);
        v16bf A1l = load_frag_a(a1l + k);
#pragma unroll
        for (int j = 0; j < 4; ++j) {
            v16bf Bh = load_frag_b(bHiP + (size_t)(16 * j) * DD + k);
            v16bf Bl = load_frag_b(bLoP + (size_t)(16 * j) * DD + k);
            acc0[j] = __builtin_amdgcn_wmma_f32_16x16x32_bf16(false, A0l, false, Bh, (short)0, acc0[j], false, false);
            acc0[j] = __builtin_amdgcn_wmma_f32_16x16x32_bf16(false, A0h, false, Bl, (short)0, acc0[j], false, false);
            acc0[j] = __builtin_amdgcn_wmma_f32_16x16x32_bf16(false, A0h, false, Bh, (short)0, acc0[j], false, false);
            acc1[j] = __builtin_amdgcn_wmma_f32_16x16x32_bf16(false, A1l, false, Bh, (short)0, acc1[j], false, false);
            acc1[j] = __builtin_amdgcn_wmma_f32_16x16x32_bf16(false, A1h, false, Bl, (short)0, acc1[j], false, false);
            acc1[j] = __builtin_amdgcn_wmma_f32_16x16x32_bf16(false, A1h, false, Bh, (short)0, acc1[j], false, false);
        }
    }

    // Epilogue: C/D layout -> element [M = rr + 8*half][N = r]
    const int mOff = half * 8;
#pragma unroll
    for (int j = 0; j < 4; ++j) {
        const size_t col = (size_t)h0 + 16 * j + r;
#pragma unroll
        for (int rr = 0; rr < 8; ++rr) {
            float c0 = acc0[j][rr];
            float c1 = acc1[j][rr];
            float d0 = fmaxf(2.0f - 2.0f * c0, 0.0f);
            float d1 = fmaxf(2.0f - 2.0f * c1, 0.0f);
            latent[(size_t)(m0 + mOff + rr)      * HH + col] = 2.0f - sqrtf(d0);
            latent[(size_t)(m0 + 16 + mOff + rr) * HH + col] = 2.0f - sqrtf(d1);
        }
    }
}

// ---------------------------------------------------------------------------
// Kernel 4: exact (topk+1)-th-largest per row via 3-pass radix select on IEEE
// bits (sim >= 0, so uint order == float order), then strict-greater mask.
// ---------------------------------------------------------------------------
__global__ __launch_bounds__(256) void topk_mask_kernel(
    float* __restrict__ latent_base, const int* __restrict__ topk_ptr)
{
    __shared__ unsigned hist[2048];
    __shared__ unsigned sh_sel;
    __shared__ int sh_k;
    float* row = latent_base + (size_t)blockIdx.y * NN * HH + (size_t)blockIdx.x * HH;
    int kk = topk_ptr[0] + 1;     // (topk+1)-th largest == kthvalue threshold
    unsigned prefix = 0;

    for (int pass = 0; pass < 3; ++pass) {
        const int shift = (pass == 0) ? 21 : (pass == 1) ? 10 : 0;
        const int bits  = (pass == 2) ? 10 : 11;
        const int nb    = 1 << bits;
        for (int i = threadIdx.x; i < nb; i += 256) hist[i] = 0;
        __syncthreads();
        for (int i = threadIdx.x; i < HH; i += 256) {
            unsigned u = __float_as_uint(row[i]);
            if (pass == 0 || (u >> (shift + bits)) == prefix)
                atomicAdd(&hist[(u >> shift) & (unsigned)(nb - 1)], 1u);
        }
        __syncthreads();
        if (threadIdx.x == 0) {
            int cum = 0; unsigned sel = 0; int krem = kk;
            for (int b = nb - 1; b >= 0; --b) {
                int c = (int)hist[b];
                if (cum + c >= kk) { sel = (unsigned)b; krem = kk - cum; break; }
                cum += c;
            }
            sh_sel = sel; sh_k = krem;
        }
        __syncthreads();
        prefix = (prefix << bits) | sh_sel;
        kk = sh_k;
        __syncthreads();
    }

    const float thres = __uint_as_float(prefix);   // exact (topk+1)-th largest
    for (int i = threadIdx.x; i < HH; i += 256) {
        float v = row[i];
        row[i] = (v > thres) ? v : 0.0f;
    }
}

// ---------------------------------------------------------------------------
// Kernel 5: sparse decode. Gather <=64 nonzeros of a latent row into LDS,
// then recon[n,:] = bias + sum_j val_j * W^T[h_j, :] with coalesced rows.
// ---------------------------------------------------------------------------
__global__ __launch_bounds__(256) void decoder_kernel(
    const float* __restrict__ latent_base,  // [2][N][H] (masked)
    const float* __restrict__ WvT,          // [H, D]
    const float* __restrict__ WtT,          // [H, D]
    const float* __restrict__ bv, const float* __restrict__ bt,
    float* __restrict__ recon_base)         // [2][N][D]
{
    __shared__ int   sh_idx[64];
    __shared__ float sh_val[64];
    __shared__ int   sh_cnt;
    const int n = blockIdx.x, m = blockIdx.y;
    const float* row  = latent_base + (size_t)m * NN * HH + (size_t)n * HH;
    const float* W    = m ? WtT : WvT;
    const float* bias = m ? bt  : bv;
    float* out = recon_base + (size_t)m * NN * DD + (size_t)n * DD;

    if (threadIdx.x == 0) sh_cnt = 0;
    __syncthreads();
    for (int i = threadIdx.x; i < HH; i += 256) {
        float v = row[i];
        if (v != 0.0f) {
            int p = atomicAdd(&sh_cnt, 1);
            if (p < 64) { sh_idx[p] = i; sh_val[p] = v; }
        }
    }
    __syncthreads();
    const int cnt = (sh_cnt < 64) ? sh_cnt : 64;

    for (int d = threadIdx.x; d < DD; d += 256) {
        float acc = bias[d];
        for (int j = 0; j < cnt; ++j)
            acc += sh_val[j] * W[(size_t)sh_idx[j] * DD + d];
        out[d] = acc;
    }
}

// ---------------------------------------------------------------------------
extern "C" void kernel_launch(void* const* d_in, const int* in_sizes, int n_in,
                              void* d_out, int out_size, void* d_ws, size_t ws_size,
                              hipStream_t stream) {
    const float* vis = (const float*)d_in[0];
    const float* txt = (const float*)d_in[1];
    const float* enc = (const float*)d_in[2];
    const float* Wv  = (const float*)d_in[3];
    const float* bv  = (const float*)d_in[4];
    const float* Wt  = (const float*)d_in[5];
    const float* bt  = (const float*)d_in[6];
    const int* topk  = (const int*)d_in[7];

    // Workspace layout (~168 MiB total)
    char* ws = (char*)d_ws;
    unsigned short* wHi  = (unsigned short*)ws; ws += (size_t)HH * DD * 2;
    unsigned short* wLo  = (unsigned short*)ws; ws += (size_t)HH * DD * 2;
    unsigned short* eVhi = (unsigned short*)ws; ws += (size_t)NN * DD * 2;
    unsigned short* eVlo = (unsigned short*)ws; ws += (size_t)NN * DD * 2;
    unsigned short* eThi = (unsigned short*)ws; ws += (size_t)NN * DD * 2;
    unsigned short* eTlo = (unsigned short*)ws; ws += (size_t)NN * DD * 2;
    float* WvT = (float*)ws; ws += (size_t)HH * DD * 4;
    float* WtT = (float*)ws; ws += (size_t)HH * DD * 4;

    // Output layout: [recon_v N*D][recon_t N*D][latent_v N*H][latent_t N*H]
    float* out         = (float*)d_out;
    float* recon_base  = out;
    float* latent_base = out + (size_t)2 * NN * DD;

    // 1) normalize + bf16 hi/lo split
    l2norm_split_kernel<<<HH, 256, 0, stream>>>(enc, wHi, wLo, DD);
    l2norm_split_kernel<<<NN, 256, 0, stream>>>(vis, eVhi, eVlo, DD);
    l2norm_split_kernel<<<NN, 256, 0, stream>>>(txt, eThi, eTlo, DD);

    // 2) transpose decoder weights for coalesced sparse decode
    transpose_kernel<<<dim3(HH / 32, DD / 32), 256, 0, stream>>>(Wv, WvT, DD, HH);
    transpose_kernel<<<dim3(HH / 32, DD / 32), 256, 0, stream>>>(Wt, WtT, DD, HH);

    // 3) encode GEMM + activation -> latent (sim) via WMMA bf16x3, 32x64 tiles
    sim_gemm_kernel<<<dim3(HH / 512, NN / 32, 2), 256, 0, stream>>>(
        eVhi, eVlo, eThi, eTlo, wHi, wLo, latent_base);

    // 4) exact top-(k+1) threshold + strict mask, in place
    topk_mask_kernel<<<dim3(NN, 2), 256, 0, stream>>>(latent_base, topk);

    // 5) sparse decode -> recon
    decoder_kernel<<<dim3(NN, 2), 256, 0, stream>>>(
        latent_base, WvT, WtT, bv, bt, recon_base);
}
